// MyQuantize_13408887898751
// MI455X (gfx1250) — matrix operand order repaired and checked by reading, not compiled, not run on hardware
//
#include <hip/hip_runtime.h>
#include <hip/hip_bf16.h>

// ---------------------------------------------------------------------------
// VQ-VAE quantize + EMA codebook update for MI455X (gfx1250, wave32, WMMA)
// Round 3: fix async builtin pointer types (v4i AS1 src, v4i AS3 dst).
// ---------------------------------------------------------------------------

typedef __attribute__((ext_vector_type(16))) __bf16 v16bf;
typedef __attribute__((ext_vector_type(8)))  float  v8f;
typedef __attribute__((ext_vector_type(4)))  int    v4i;

#define VQ_N      131072      // 8*16*32*32 rows
#define VQ_D      256         // dim
#define VQ_NE     1024        // n_embed
#define VQ_DECAY  0.99f
#define VQ_ALPHA  0.01f       // 1 - DECAY
#define VQ_EPS    1e-5f
#define AST       264         // padded LDS row stride for A (256 + 8 bf16)
#define TILE_USH  (VQ_D * 16) // 4096 ushorts = 8192 bytes per B tile

__device__ __forceinline__ unsigned short f2bf(float f) {
  // round-to-nearest-even float -> bf16 (bit trick; NaN not special-cased)
  unsigned int u = __float_as_uint(f);
  u += 0x7FFFu + ((u >> 16) & 1u);
  return (unsigned short)(u >> 16);
}

// ---- CDNA5 async global->LDS helpers (guarded; fallback keeps compile OK) --
__device__ __forceinline__ void async_tile_load(const unsigned short* gsrc,
                                                unsigned short* ldst, int tid) {
  // 8192 bytes: 128 threads x 4 x b128
#if __has_builtin(__builtin_amdgcn_global_load_async_to_lds_b128)
  typedef __attribute__((address_space(1))) v4i gv4i;
  typedef __attribute__((address_space(3))) v4i lv4i;
  #pragma unroll
  for (int i = 0; i < 4; ++i) {
    __builtin_amdgcn_global_load_async_to_lds_b128(
        (gv4i*)(gsrc + tid * 8 + i * 1024),
        (lv4i*)(ldst + tid * 8 + i * 1024),
        /*offset=*/0, /*cpol=*/0);
  }
#else
  #pragma unroll
  for (int i = 0; i < 4; ++i) {
    *(uint4*)(ldst + tid * 8 + i * 1024) =
        *(const uint4*)(gsrc + tid * 8 + i * 1024);
  }
#endif
}

__device__ __forceinline__ void wait_async0() {
#if __has_builtin(__builtin_amdgcn_s_wait_asynccnt)
  __builtin_amdgcn_s_wait_asynccnt(0);
#else
  asm volatile("s_wait_asynccnt 0" ::: "memory");
#endif
}

// --------------------------- 1) zero accumulators ---------------------------
__global__ __launch_bounds__(256)
void vq_init(float* __restrict__ esum, float* __restrict__ oh,
             float* __restrict__ misc) {
  int i = blockIdx.x * 256 + threadIdx.x;
  if (i < VQ_D * VQ_NE) esum[i] = 0.0f;
  if (i < VQ_NE)        oh[i]   = 0.0f;
  if (i < 2)            misc[i] = 0.0f;
}

// ------------- 2) codebook column norms + bf16 tiled repack ------------------
__global__ __launch_bounds__(256)
void vq_norme(const float* __restrict__ embed, float* __restrict__ norme) {
  int j = blockIdx.x * 256 + threadIdx.x;   // 0..1023, coalesced over j
  float s = 0.0f;
  #pragma unroll 4
  for (int d = 0; d < VQ_D; ++d) {
    float v = embed[d * VQ_NE + j];
    s += v * v;
  }
  norme[j] = s;
}

// Repack embed[d,j] (fp32, j fastest) -> bf16 tiles [ct][n][k], each tile a
// contiguous 8KB block so the WMMA kernel can DMA it with async b128 loads.
__global__ __launch_bounds__(256)
void vq_embbf(const float* __restrict__ embed, unsigned short* __restrict__ embbf) {
  int e = blockIdx.x * 256 + threadIdx.x;   // e = d*1024 + j
  int d = e >> 10, j = e & 1023;
  int ct = j >> 4, n = j & 15;
  embbf[ct * TILE_USH + n * VQ_D + d] = f2bf(embed[e]);
}

// --------------------------- 3) WMMA distance + argmin -----------------------
// Block: 128 threads = 4 waves, 16 rows per wave -> 64 rows per block.
// Score = ||E_j||^2 - 2 * <x_i, E_j>  (row-constant ||x||^2 dropped: argmin-safe)
__global__ __launch_bounds__(128)
void vq_argmin(const float* __restrict__ input,
               const unsigned short* __restrict__ embbf,
               const float* __restrict__ norme, int* __restrict__ ind,
               float* __restrict__ out_ind) {
  __shared__ unsigned short sA[64 * AST];                  // 64 rows x 256 K bf16
  __shared__ __align__(16) unsigned short sB[2][TILE_USH]; // double-buffered B
  __shared__ float          sNE[VQ_NE];                    // ||E_j||^2

  const int tid     = threadIdx.x;
  const int lane    = tid & 31;
  const int wave    = tid >> 5;               // 0..3
  const int rowBase = blockIdx.x * 64;
  const int m       = lane & 15;              // row-in-tile / column-in-tile id
  const int half    = lane >> 4;              // lane half selects K sub-block

  // Kick off DMA of B tile 0 while we stage A.
  async_tile_load(embbf, sB[0], tid);

  // Stage A tile (fp32 -> bf16), fully coalesced.
  for (int e = tid; e < 64 * VQ_D; e += 128) {
    int r = e >> 8, k = e & 255;
    sA[r * AST + k] = f2bf(input[(long long)(rowBase + r) * VQ_D + k]);
  }
  for (int t = tid; t < VQ_NE; t += 128) sNE[t] = norme[t];
  __syncthreads();

  // Hoist all 8 A fragments (K=256 -> 8 k-steps of 32) into VGPRs.
  // ISA 16-bit A layout: lanes 0-15 hold K {0..7,16..23}, lanes 16-31 hold
  // K {8..15,24..31} -> two contiguous 16B LDS reads per fragment.
  union Frag { v16bf v; uint4 q[2]; };
  Frag afrag[8];
  {
    const unsigned short* rowp = &sA[(wave * 16 + m) * AST];
    #pragma unroll
    for (int kk = 0; kk < 8; ++kk) {
      afrag[kk].q[0] = *(const uint4*)(rowp + kk * 32 + half * 8);
      afrag[kk].q[1] = *(const uint4*)(rowp + kk * 32 + 16 + half * 8);
    }
  }

  float best[8];
  int   bidx[8];
  #pragma unroll
  for (int v = 0; v < 8; ++v) { best[v] = 3.4e38f; bidx[v] = 0; }

  for (int ct = 0; ct < VQ_NE / 16; ++ct) {
    // Tile ct's DMA was issued last iteration (or in the prologue).
    wait_async0();
    __syncthreads();   // all waves' async chunks visible; prev buffer reads done

    if (ct + 1 < VQ_NE / 16)
      async_tile_load(embbf + (ct + 1) * TILE_USH, sB[(ct + 1) & 1], tid);

    const unsigned short* bbuf = sB[ct & 1];
    v8f c = {};
    #pragma unroll
    for (int kk = 0; kk < 8; ++kk) {
      // B layout: lane l holds fixed N = m, K {0..7,16..23} (+8 for half 1)
      // -> two contiguous 16B LDS reads.
      Frag bf;
      const unsigned short* bp = &bbuf[m * VQ_D + kk * 32 + half * 8];
      bf.q[0] = *(const uint4*)bp;
      bf.q[1] = *(const uint4*)(bp + 16);
      c = __builtin_amdgcn_wmma_f32_16x16x32_bf16(
              /*neg_a=*/false, afrag[kk].v, /*neg_b=*/false, bf.v,
              /*c_mod=*/(short)0, c, /*reuse_a=*/false, /*reuse_b=*/false);
    }

    // C layout: VGPR v -> row (v or v+8 by lane half), lane&15 -> column.
    const int   j  = ct * 16 + m;
    const float ne = sNE[j];
    #pragma unroll
    for (int v = 0; v < 8; ++v) {
      float score = ne - 2.0f * c[v];
      if (score < best[v]) { best[v] = score; bidx[v] = j; }
    }
  }

  // Reduce min/argmin across the 16 lanes of each half-wave (first-index ties).
  #pragma unroll
  for (int msk = 1; msk <= 8; msk <<= 1) {
    #pragma unroll
    for (int v = 0; v < 8; ++v) {
      float os = __shfl_xor(best[v], msk, 32);
      int   oi = __shfl_xor(bidx[v], msk, 32);
      if (os < best[v] || (os == best[v] && oi < bidx[v])) {
        best[v] = os; bidx[v] = oi;
      }
    }
  }
  if (m == 0) {
    #pragma unroll
    for (int v = 0; v < 8; ++v) {
      int row = rowBase + wave * 16 + half * 8 + v;   // half 0: rows 0-7, half 1: 8-15
      ind[row]     = bidx[v];
      out_ind[row] = (float)bidx[v];
    }
  }
}

// ------------------- 4) gather + diff + segment scatter-add ------------------
__global__ __launch_bounds__(256)
void vq_gather(const float* __restrict__ input, const float* __restrict__ embed,
               const int* __restrict__ ind, float* __restrict__ out_q,
               float* __restrict__ esum, float* __restrict__ oh,
               float* __restrict__ diffacc) {
  __shared__ float red[256];
  const int row = blockIdx.x;
  const int t   = threadIdx.x;
  const int j   = ind[row];
  const float x = input[(long long)row * VQ_D + t];
  const float q = embed[t * VQ_NE + j];            // codebook is L2-resident (1 MB)
  out_q[(long long)row * VQ_D + t] = x + (q - x);  // straight-through value
  float d = q - x;
  red[t] = d * d;
  unsafeAtomicAdd(&esum[t * VQ_NE + j], x);        // global_atomic_add_f32
  __syncthreads();
  for (int s = 128; s > 0; s >>= 1) {
    if (t < s) red[t] += red[t + s];
    __syncthreads();
  }
  if (t == 0) {
    unsafeAtomicAdd(diffacc, red[0]);
    unsafeAtomicAdd(&oh[j], 1.0f);
  }
}

// ------------------- 5) EMA cluster size, n, diff finalize -------------------
__global__ __launch_bounds__(1024)
void vq_cs(const float* __restrict__ cluster_size, const float* __restrict__ oh,
           const float* __restrict__ diffacc, float* __restrict__ out_cs,
           float* __restrict__ out_diff, float* __restrict__ nval) {
  __shared__ float red[1024];
  int t = threadIdx.x;
  float ncs = cluster_size[t] * VQ_DECAY + VQ_ALPHA * oh[t];
  out_cs[t] = ncs;
  red[t] = ncs;
  __syncthreads();
  for (int s = 512; s > 0; s >>= 1) {
    if (t < s) red[t] += red[t + s];
    __syncthreads();
  }
  if (t == 0) {
    nval[0] = red[0];
    out_diff[0] = diffacc[0] / (float)((long long)VQ_N * VQ_D);
  }
}

// ------------------- 6) EMA embed_avg and new_embed --------------------------
__global__ __launch_bounds__(256)
void vq_embed(const float* __restrict__ embed_avg, const float* __restrict__ esum,
              const float* __restrict__ out_cs, const float* __restrict__ nval,
              float* __restrict__ out_avg, float* __restrict__ out_embed) {
  int e = blockIdx.x * 256 + threadIdx.x;       // 0 .. D*NE-1, j fastest
  int j = e & (VQ_NE - 1);
  float n   = nval[0];
  float ncs = out_cs[j];
  float cs  = (ncs + VQ_EPS) / (n + VQ_NE * VQ_EPS) * n;
  float avg = embed_avg[e] * VQ_DECAY + VQ_ALPHA * esum[e];
  out_avg[e]   = avg;
  out_embed[e] = avg / cs;
}

// ---------------------------------------------------------------------------
extern "C" void kernel_launch(void* const* d_in, const int* in_sizes, int n_in,
                              void* d_out, int out_size, void* d_ws, size_t ws_size,
                              hipStream_t stream) {
  (void)in_sizes; (void)n_in; (void)out_size; (void)ws_size;

  const float* input        = (const float*)d_in[0]; // [N, 256]
  const float* embed        = (const float*)d_in[1]; // [256, 1024]
  const float* cluster_size = (const float*)d_in[2]; // [1024]
  const float* embed_avg    = (const float*)d_in[3]; // [256, 1024]

  // Outputs, concatenated flat in reference return order.
  float* out      = (float*)d_out;
  float* out_q    = out;                                  // N*D
  float* out_diff = out_q + (long long)VQ_N * VQ_D;       // 1
  float* out_ind  = out_diff + 1;                         // N
  float* out_emb  = out_ind + VQ_N;                       // D*NE
  float* out_cs   = out_emb + (long long)VQ_D * VQ_NE;    // NE
  float* out_avg  = out_cs + VQ_NE;                       // D*NE

  // Workspace layout (all 16B-aligned blocks).
  char*  wsb      = (char*)d_ws;
  int*   ws_ind   = (int*)wsb;                                   // N ints (512KB)
  float* ws_norme = (float*)(wsb + sizeof(int) * (size_t)VQ_N);  // NE
  float* ws_esum  = ws_norme + VQ_NE;                            // D*NE (1MB)
  float* ws_oh    = ws_esum + (long long)VQ_D * VQ_NE;           // NE
  unsigned short* ws_embbf = (unsigned short*)(ws_oh + VQ_NE);   // D*NE bf16 (512KB)
  float* ws_misc  = (float*)(ws_embbf + (long long)VQ_D * VQ_NE); // [0]=diff, [1]=n

  vq_init  <<<(VQ_D * VQ_NE + 255) / 256, 256, 0, stream>>>(ws_esum, ws_oh, ws_misc);
  vq_norme <<<VQ_NE / 256, 256, 0, stream>>>(embed, ws_norme);
  vq_embbf <<<(VQ_D * VQ_NE) / 256, 256, 0, stream>>>(embed, ws_embbf);
  vq_argmin<<<VQ_N / 64, 128, 0, stream>>>(input, ws_embbf, ws_norme, ws_ind, out_ind);
  vq_gather<<<VQ_N, 256, 0, stream>>>(input, embed, ws_ind, out_q, ws_esum, ws_oh,
                                      &ws_misc[0]);
  vq_cs    <<<1, 1024, 0, stream>>>(cluster_size, ws_oh, &ws_misc[0], out_cs,
                                    out_diff, &ws_misc[1]);
  vq_embed <<<(VQ_D * VQ_NE) / 256, 256, 0, stream>>>(embed_avg, ws_esum, out_cs,
                                                      &ws_misc[1], out_avg, out_emb);
}
